// globalAttention_15281493639726
// MI455X (gfx1250) — matrix operand order repaired and checked by reading, not compiled, not run on hardware
//
#include <hip/hip_runtime.h>

typedef __bf16 bf16_t;
typedef __attribute__((ext_vector_type(16))) __bf16 v16bf;
typedef __attribute__((ext_vector_type(8)))  float  v8f;

#define NFEAT 64
#define NTOK  1024
#define DDIM  4096

// ---------------------------------------------------------------------------
// K1: depthwise 3x3 convs for q and k -> bf16 token matrices
//   Qt/Kt : [b][token=1024][d=4096] row-major.
// ---------------------------------------------------------------------------
__global__ __launch_bounds__(256) void k_qk_dw(
    const float* __restrict__ x,
    const float* __restrict__ qw, const float* __restrict__ qb,
    const float* __restrict__ kw, const float* __restrict__ kb,
    bf16_t* __restrict__ Qt, bf16_t* __restrict__ Kt)
{
  int idx = blockIdx.x * 256 + threadIdx.x;   // 64 frames * 64 ch * 4096 px
  int px  = idx & 4095;
  int fc  = idx >> 12;
  int c   = fc & 63;
  int f   = fc >> 6;
  int y   = px >> 6, xx = px & 63;
  const float* xin = x + (((size_t)f * NFEAT + c) << 12);
  float qa = qb[c], ka = kb[c];
#pragma unroll
  for (int dy = -1; dy <= 1; ++dy) {
    int yy = y + dy;
    if ((unsigned)yy >= 64u) continue;
#pragma unroll
    for (int dx = -1; dx <= 1; ++dx) {
      int xc = xx + dx;
      if ((unsigned)xc >= 64u) continue;
      float v = xin[yy * 64 + xc];
      int  wi = c * 9 + (dy + 1) * 3 + (dx + 1);
      qa += qw[wi] * v;
      ka += kw[wi] * v;
    }
  }
  int bb = f >> 4, t = f & 15;
  int j = t * 64 + (y >> 3) * 8 + (xx >> 3);
  int d = c * 64 + (y & 7) * 8 + (xx & 7);
  size_t o = (((size_t)bb * NTOK) + j) * DDIM + d;
  Qt[o] = (bf16_t)qa;
  Kt[o] = (bf16_t)ka;
}

// ---------------------------------------------------------------------------
// K2: full 3x3 conv for v, LDS-tiled, written TRANSPOSED:
//   VT : [b][d=4096][token=1024]  (WMMA B columns contiguous along tokens)
// ---------------------------------------------------------------------------
__global__ __launch_bounds__(256) void k_vconv(
    const float* __restrict__ x,
    const float* __restrict__ vw, const float* __restrict__ vb,
    bf16_t* __restrict__ VT)
{
  __shared__ float lds[NFEAT][10][10];
  int f = blockIdx.x, patch = blockIdx.y;
  int gh = patch >> 3, gw = patch & 7;
  int y0 = gh * 8 - 1, x0 = gw * 8 - 1;
  const float* xin = x + ((size_t)f * NFEAT << 12);
  for (int e = threadIdx.x; e < NFEAT * 100; e += 256) {
    int c = e / 100, r = e % 100;
    int ry = r / 10, rx = r % 10;
    int yy = y0 + ry, xc = x0 + rx;
    float v = 0.f;
    if ((unsigned)yy < 64u && (unsigned)xc < 64u) v = xin[(c << 12) + yy * 64 + xc];
    lds[c][ry][rx] = v;
  }
  __syncthreads();
  int oc  = threadIdx.x & 63;
  int grp = threadIdx.x >> 6;
  float acc[16];
  float bias = vb[oc];
#pragma unroll
  for (int p = 0; p < 16; ++p) acc[p] = bias;
  const float* w = vw + (size_t)oc * (NFEAT * 9);
  for (int i = 0; i < NFEAT; ++i) {
    float w0 = w[i*9+0], w1 = w[i*9+1], w2 = w[i*9+2];
    float w3 = w[i*9+3], w4 = w[i*9+4], w5 = w[i*9+5];
    float w6 = w[i*9+6], w7 = w[i*9+7], w8 = w[i*9+8];
#pragma unroll
    for (int p = 0; p < 16; ++p) {
      int px = grp * 16 + p;
      int py = px >> 3, pxx = px & 7;
      acc[p] += w0*lds[i][py  ][pxx] + w1*lds[i][py  ][pxx+1] + w2*lds[i][py  ][pxx+2]
              + w3*lds[i][py+1][pxx] + w4*lds[i][py+1][pxx+1] + w5*lds[i][py+1][pxx+2]
              + w6*lds[i][py+2][pxx] + w7*lds[i][py+2][pxx+1] + w8*lds[i][py+2][pxx+2];
    }
  }
  int bb = f >> 4, t = f & 15;
  int j = t * 64 + patch;
#pragma unroll
  for (int p = 0; p < 16; ++p) {
    int px = grp * 16 + p;
    int py = px >> 3, pxx = px & 7;
    int d = oc * 64 + py * 8 + pxx;
    VT[(((size_t)bb * DDIM) + d) * NTOK + j] = (bf16_t)acc[p];
  }
}

// ---------------------------------------------------------------------------
// Async staging helper: one b128 (16B) chunk  global -> LDS  via ASYNCcnt.
// ldsoff = LDS byte offset (low 32 bits of the flat shared-memory address).
// ---------------------------------------------------------------------------
__device__ __forceinline__ void async_copy16(unsigned ldsoff, const bf16_t* g)
{
  asm volatile("global_load_async_to_lds_b128 %0, %1, off"
               :: "v"(ldsoff), "v"(g) : "memory");
}
__device__ __forceinline__ void wait_async0()
{
  asm volatile("s_wait_asynccnt 0x0" ::: "memory");
}

// ---------------------------------------------------------------------------
// K3a: S = (Q^T K)/sqrt(d) via WMMA bf16 with double-buffered async-LDS
// staging. Block = 4 waves computes a 64x64 tile of S; per K-step (32) the
// block stages A(64x32) and B(64x32) bf16 tiles into LDS with
// GLOBAL_LOAD_ASYNC_TO_LDS_B128, overlapping next-step loads with WMMA.
// ---------------------------------------------------------------------------
__global__ __launch_bounds__(128) void k_attn_qk(
    const bf16_t* __restrict__ Qt, const bf16_t* __restrict__ Kt,
    float* __restrict__ S)
{
  __shared__ bf16_t As[2][64][32];   // 8 KB
  __shared__ bf16_t Bs[2][64][32];   // 8 KB

  int lane = threadIdx.x;
  int wave = threadIdx.y;
  int tid  = wave * 32 + lane;
  int b    = blockIdx.z;
  int half = lane >> 4, l = lane & 15;
  int tileI = blockIdx.y * 64;
  int tileJ = blockIdx.x * 64;

  const bf16_t* Qbase = Qt + (((size_t)b * NTOK) + tileI) * DDIM;
  const bf16_t* Kbase = Kt + (((size_t)b * NTOK) + tileJ) * DDIM;

  // per-thread staging assignment: 2 x (A chunk + B chunk) of 16B per step
  int row0 = (tid      ) >> 1;        // 0..63
  int ko0  = ((tid      ) & 1) * 8;   // bf16 elements {0,8}
  int row1 = (tid + 128) >> 1;
  int ko1  = ((tid + 128) & 1) * 8;   // covers k elements 16,24

  v8f a0{}, a1{}, a2{}, a3{};

  // prologue: stage kk = 0 into buffer 0
  {
    async_copy16((unsigned)(uintptr_t)&As[0][row0][ko0], Qbase + (size_t)row0 * DDIM + ko0);
    async_copy16((unsigned)(uintptr_t)&Bs[0][row0][ko0], Kbase + (size_t)row0 * DDIM + ko0);
    async_copy16((unsigned)(uintptr_t)&As[0][row1][ko1 + 16], Qbase + (size_t)row1 * DDIM + ko1 + 16);
    async_copy16((unsigned)(uintptr_t)&Bs[0][row1][ko1 + 16], Kbase + (size_t)row1 * DDIM + ko1 + 16);
  }

  int buf = 0;
  for (int kk = 0; kk < DDIM; kk += 32) {
    wait_async0();
    __syncthreads();
    if (kk + 32 < DDIM) {
      int nb = buf ^ 1, nk = kk + 32;
      async_copy16((unsigned)(uintptr_t)&As[nb][row0][ko0], Qbase + (size_t)row0 * DDIM + nk + ko0);
      async_copy16((unsigned)(uintptr_t)&Bs[nb][row0][ko0], Kbase + (size_t)row0 * DDIM + nk + ko0);
      async_copy16((unsigned)(uintptr_t)&As[nb][row1][ko1 + 16], Qbase + (size_t)row1 * DDIM + nk + ko1 + 16);
      async_copy16((unsigned)(uintptr_t)&Bs[nb][row1][ko1 + 16], Kbase + (size_t)row1 * DDIM + nk + ko1 + 16);
    }
    v16bf af;
    af = *(const v16bf*)&As[buf][wave * 16 + l][half * 16];
    v16bf b0 = *(const v16bf*)&Bs[buf][ 0 + l][half * 16];
    v16bf b1 = *(const v16bf*)&Bs[buf][16 + l][half * 16];
    v16bf b2 = *(const v16bf*)&Bs[buf][32 + l][half * 16];
    v16bf b3 = *(const v16bf*)&Bs[buf][48 + l][half * 16];
    a0 = __builtin_amdgcn_wmma_f32_16x16x32_bf16(false, af, false, b0, (short)0, a0, false, false);
    a1 = __builtin_amdgcn_wmma_f32_16x16x32_bf16(false, af, false, b1, (short)0, a1, false, false);
    a2 = __builtin_amdgcn_wmma_f32_16x16x32_bf16(false, af, false, b2, (short)0, a2, false, false);
    a3 = __builtin_amdgcn_wmma_f32_16x16x32_bf16(false, af, false, b3, (short)0, a3, false, false);
    buf ^= 1;
  }

  const float scale = 0.015625f;   // 4096^-0.5
#pragma unroll
  for (int r = 0; r < 8; ++r) {
    int i = tileI + wave * 16 + r + half * 8;
    float* srow = S + (((size_t)b * NTOK) + i) * NTOK + tileJ + l;
    srow[ 0] = a0[r] * scale;
    srow[16] = a1[r] * scale;
    srow[32] = a2[r] * scale;
    srow[48] = a3[r] * scale;
  }
}

// ---------------------------------------------------------------------------
// K3b: row softmax over S (length 1024); row converted in place to bf16
// probabilities at the row start (row stride stays 4096 bytes).
// ---------------------------------------------------------------------------
__global__ __launch_bounds__(256) void k_softmax(float* __restrict__ S)
{
  __shared__ float red[8];
  size_t row = blockIdx.x;            // b*1024 + i, grid = 4096
  float* s = S + row * NTOK;
  int tid  = threadIdx.x;
  int lane = tid & 31;
  float v[4];
  float m = -3.0e38f;
#pragma unroll
  for (int q = 0; q < 4; ++q) { v[q] = s[tid + q * 256]; m = fmaxf(m, v[q]); }
#pragma unroll
  for (int o = 16; o > 0; o >>= 1) m = fmaxf(m, __shfl_xor(m, o));
  if (lane == 0) red[tid >> 5] = m;
  __syncthreads();
  float mg = red[0];
#pragma unroll
  for (int w = 1; w < 8; ++w) mg = fmaxf(mg, red[w]);
  float sum = 0.f;
#pragma unroll
  for (int q = 0; q < 4; ++q) { v[q] = __expf(v[q] - mg); sum += v[q]; }
#pragma unroll
  for (int o = 16; o > 0; o >>= 1) sum += __shfl_xor(sum, o);
  __syncthreads();
  if (lane == 0) red[tid >> 5] = sum;
  __syncthreads();
  float tot = 0.f;
#pragma unroll
  for (int w = 0; w < 8; ++w) tot += red[w];
  float inv = 1.0f / tot;
  bf16_t* p = (bf16_t*)s;             // in place: all reads happened pre-barrier
#pragma unroll
  for (int q = 0; q < 4; ++q) p[tid + q * 256] = (bf16_t)(v[q] * inv);
}

// ---------------------------------------------------------------------------
// K3c: O = P * V^T via WMMA bf16, same double-buffered async-LDS scheme
// (K-dim = 1024 tokens). Results scattered directly into fold() layout
// feat[frame][c][y][x].
// ---------------------------------------------------------------------------
__global__ __launch_bounds__(128) void k_attn_pv(
    const float* __restrict__ Sbase, const bf16_t* __restrict__ VT,
    float* __restrict__ feat)
{
  __shared__ bf16_t As[2][64][32];
  __shared__ bf16_t Bs[2][64][32];

  int lane = threadIdx.x;
  int wave = threadIdx.y;
  int tid  = wave * 32 + lane;
  int b    = blockIdx.z;
  int half = lane >> 4, l = lane & 15;
  int tileI = blockIdx.y * 64;   // token dim (rows of P)
  int tileD = blockIdx.x * 64;   // d dim (cols)

  // P rows: bf16 data at start of each fp32 S row -> row stride 2048 bf16
  const bf16_t* Pbase = (const bf16_t*)(Sbase + (((size_t)b * NTOK) + tileI) * NTOK);
  const bf16_t* Vbase = VT + (((size_t)b * DDIM) + tileD) * NTOK;

  int row0 = (tid      ) >> 1;
  int ko0  = ((tid      ) & 1) * 8;
  int row1 = (tid + 128) >> 1;
  int ko1  = ((tid + 128) & 1) * 8;

  v8f a0{}, a1{}, a2{}, a3{};

  {
    async_copy16((unsigned)(uintptr_t)&As[0][row0][ko0], Pbase + (size_t)row0 * 2048 + ko0);
    async_copy16((unsigned)(uintptr_t)&Bs[0][row0][ko0], Vbase + (size_t)row0 * NTOK + ko0);
    async_copy16((unsigned)(uintptr_t)&As[0][row1][ko1 + 16], Pbase + (size_t)row1 * 2048 + ko1 + 16);
    async_copy16((unsigned)(uintptr_t)&Bs[0][row1][ko1 + 16], Vbase + (size_t)row1 * NTOK + ko1 + 16);
  }

  int buf = 0;
  for (int kk = 0; kk < NTOK; kk += 32) {
    wait_async0();
    __syncthreads();
    if (kk + 32 < NTOK) {
      int nb = buf ^ 1, nk = kk + 32;
      async_copy16((unsigned)(uintptr_t)&As[nb][row0][ko0], Pbase + (size_t)row0 * 2048 + nk + ko0);
      async_copy16((unsigned)(uintptr_t)&Bs[nb][row0][ko0], Vbase + (size_t)row0 * NTOK + nk + ko0);
      async_copy16((unsigned)(uintptr_t)&As[nb][row1][ko1 + 16], Pbase + (size_t)row1 * 2048 + nk + ko1 + 16);
      async_copy16((unsigned)(uintptr_t)&Bs[nb][row1][ko1 + 16], Vbase + (size_t)row1 * NTOK + nk + ko1 + 16);
    }
    v16bf af;
    af = *(const v16bf*)&As[buf][wave * 16 + l][half * 16];
    v16bf b0 = *(const v16bf*)&Bs[buf][ 0 + l][half * 16];
    v16bf b1 = *(const v16bf*)&Bs[buf][16 + l][half * 16];
    v16bf b2 = *(const v16bf*)&Bs[buf][32 + l][half * 16];
    v16bf b3 = *(const v16bf*)&Bs[buf][48 + l][half * 16];
    a0 = __builtin_amdgcn_wmma_f32_16x16x32_bf16(false, af, false, b0, (short)0, a0, false, false);
    a1 = __builtin_amdgcn_wmma_f32_16x16x32_bf16(false, af, false, b1, (short)0, a1, false, false);
    a2 = __builtin_amdgcn_wmma_f32_16x16x32_bf16(false, af, false, b2, (short)0, a2, false, false);
    a3 = __builtin_amdgcn_wmma_f32_16x16x32_bf16(false, af, false, b3, (short)0, a3, false, false);
    buf ^= 1;
  }

#pragma unroll
  for (int r = 0; r < 8; ++r) {
    int i  = tileI + wave * 16 + r + half * 8;
    int t  = i >> 6, np = i & 63;
    int gh = np >> 3, gw = np & 7;
    int f  = b * 16 + t;
#pragma unroll
    for (int nn = 0; nn < 4; ++nn) {
      int d  = tileD + nn * 16 + l;
      int c  = d >> 6, pi = (d >> 3) & 7, pj = d & 7;
      float val = (nn == 0) ? a0[r] : (nn == 1) ? a1[r] : (nn == 2) ? a2[r] : a3[r];
      feat[(((size_t)f * NFEAT) + c) * 4096 + (gh * 8 + pi) * 64 + (gw * 8 + pj)] = val;
    }
  }
}

// ---------------------------------------------------------------------------
// K4: final full 3x3 conv on feat + residual add -> out.
// ---------------------------------------------------------------------------
__global__ __launch_bounds__(256) void k_cconv(
    const float* __restrict__ feat,
    const float* __restrict__ cw, const float* __restrict__ cb,
    const float* __restrict__ xres, float* __restrict__ out)
{
  __shared__ float lds[NFEAT][10][10];
  int f = blockIdx.x, tile = blockIdx.y;
  int gh = tile >> 3, gw = tile & 7;
  int y0 = gh * 8 - 1, x0 = gw * 8 - 1;
  const float* fin = feat + ((size_t)f * NFEAT << 12);
  for (int e = threadIdx.x; e < NFEAT * 100; e += 256) {
    int c = e / 100, r = e % 100;
    int ry = r / 10, rx = r % 10;
    int yy = y0 + ry, xc = x0 + rx;
    float v = 0.f;
    if ((unsigned)yy < 64u && (unsigned)xc < 64u) v = fin[(c << 12) + yy * 64 + xc];
    lds[c][ry][rx] = v;
  }
  __syncthreads();
  int oc  = threadIdx.x & 63;
  int grp = threadIdx.x >> 6;
  float acc[16];
  float bias = cb[oc];
#pragma unroll
  for (int p = 0; p < 16; ++p) acc[p] = bias;
  const float* w = cw + (size_t)oc * (NFEAT * 9);
  for (int i = 0; i < NFEAT; ++i) {
    float w0 = w[i*9+0], w1 = w[i*9+1], w2 = w[i*9+2];
    float w3 = w[i*9+3], w4 = w[i*9+4], w5 = w[i*9+5];
    float w6 = w[i*9+6], w7 = w[i*9+7], w8 = w[i*9+8];
#pragma unroll
    for (int p = 0; p < 16; ++p) {
      int px = grp * 16 + p;
      int py = px >> 3, pxx = px & 7;
      acc[p] += w0*lds[i][py  ][pxx] + w1*lds[i][py  ][pxx+1] + w2*lds[i][py  ][pxx+2]
              + w3*lds[i][py+1][pxx] + w4*lds[i][py+1][pxx+1] + w5*lds[i][py+1][pxx+2]
              + w6*lds[i][py+2][pxx] + w7*lds[i][py+2][pxx+1] + w8*lds[i][py+2][pxx+2];
    }
  }
#pragma unroll
  for (int p = 0; p < 16; ++p) {
    int px = grp * 16 + p;
    int py = px >> 3, pxx = px & 7;
    size_t oi = (((size_t)f * NFEAT) + oc) * 4096 + (gh * 8 + py) * 64 + (gw * 8 + pxx);
    out[oi] = acc[p] + xres[oi];
  }
}

// ---------------------------------------------------------------------------
extern "C" void kernel_launch(void* const* d_in, const int* in_sizes, int n_in,
                              void* d_out, int out_size, void* d_ws, size_t ws_size,
                              hipStream_t stream)
{
  const float* x  = (const float*)d_in[0];
  const float* qw = (const float*)d_in[1];
  const float* qb = (const float*)d_in[2];
  const float* kw = (const float*)d_in[3];
  const float* kb = (const float*)d_in[4];
  const float* vw = (const float*)d_in[5];
  const float* vb = (const float*)d_in[6];
  const float* cw = (const float*)d_in[7];
  const float* cb = (const float*)d_in[8];
  float* out = (float*)d_out;

  char* ws = (char*)d_ws;
  // Workspace layout (151 MB total):
  //   [0,        33.5MB)  Qt  (bf16 tokens)  -- reused as VT after k_attn_qk
  //   [33.5MB,   67.1MB)  Kt  (bf16 tokens)
  //   [67.1MB,   83.9MB)  S   (fp32 logits; bf16 P written in place by softmax)
  //   [83.9MB,  151.0MB)  feat (fp32 fold output)
  bf16_t* Qt   = (bf16_t*)(ws);
  bf16_t* Kt   = (bf16_t*)(ws + 33554432);
  float*  S    = (float*) (ws + 67108864);
  float*  feat = (float*) (ws + 83886080);
  bf16_t* VT   = Qt;   // safe: k_vconv runs after k_attn_qk on the same stream

  k_qk_dw  <<<65536, 256, 0, stream>>>(x, qw, qb, kw, kb, Qt, Kt);
  k_attn_qk<<<dim3(16, 16, 4), dim3(32, 4), 0, stream>>>(Qt, Kt, S);
  k_softmax<<<4096, 256, 0, stream>>>(S);
  k_vconv  <<<dim3(64, 64), 256, 0, stream>>>(x, vw, vb, VT);
  k_attn_pv<<<dim3(64, 16, 4), dim3(32, 4), 0, stream>>>(S, VT, feat);
  k_cconv  <<<dim3(64, 64), 256, 0, stream>>>(feat, cw, cb, x, out);
}
